// GCN_65661460021749
// MI455X (gfx1250) — compile-verified, matrix operand order checked by reading, not run on hardware
//
#include <hip/hip_runtime.h>
#include <hip/hip_bf16.h>

// ---------------------------------------------------------------------------
// 2-layer GCN for MI455X (gfx1250, wave32).
// GEMMs: v_wmma_f32_16x16x32_f16 with pre-converted f16 operands loaded as
// aligned b128 fragments. Edge aggregation: float4 gathers + f32 atomics
// (feature matrices are L2-resident: 12.8 MB << 192 MB L2).
// ---------------------------------------------------------------------------

typedef __attribute__((ext_vector_type(16))) _Float16 v16h;
typedef __attribute__((ext_vector_type(8)))  _Float16 v8h;
typedef __attribute__((ext_vector_type(8)))  float    v8f;

// ---------------- utility kernels ----------------

__global__ void fill_f32(float* __restrict__ p, float val, long long n) {
    long long i = (long long)blockIdx.x * blockDim.x + threadIdx.x;
    long long stride = (long long)gridDim.x * blockDim.x;
    for (; i < n; i += stride) p[i] = val;
}

__global__ void f32_to_f16(const float* __restrict__ in, _Float16* __restrict__ out,
                           long long n) {
    long long i = (long long)blockIdx.x * blockDim.x + threadIdx.x;
    long long stride = (long long)gridDim.x * blockDim.x;
    for (; i < n; i += stride) out[i] = (_Float16)in[i];
}

// W[K x Nc] (f32, row-major) -> Wt[Nc x K] (f16, row-major)
__global__ void transpose_w_f16(const float* __restrict__ W, _Float16* __restrict__ Wt,
                                int K, int Nc) {
    int t = blockIdx.x * blockDim.x + threadIdx.x;
    if (t < K * Nc) {
        int n = t / K;
        int k = t % K;
        Wt[t] = (_Float16)W[k * Nc + n];
    }
}

// deg[col[e]] += 1 for every real edge (self-loop baked into fill value 1.0f)
__global__ void deg_accum(const int* __restrict__ col, float* __restrict__ deg, int E) {
    int i = blockIdx.x * blockDim.x + threadIdx.x;
    int stride = gridDim.x * blockDim.x;
    for (; i < E; i += stride) atomicAdd(&deg[col[i]], 1.0f);
}

// deg -> deg^{-1/2} in place (deg >= 1 always, self-loops included)
__global__ void make_dinv(float* __restrict__ deg, int N) {
    int i = blockIdx.x * blockDim.x + threadIdx.x;
    if (i < N) deg[i] = rsqrtf(deg[i]);
}

// ---------------- WMMA fragment loads (all 16B-aligned b128) ----------------
// A: 16x32 f16 tile of row-major Xh[M x 64]. Lanes 0-15 -> M=lane,
// K groups [kk..kk+7] & [kk+16..kk+23]; lanes 16-31 -> M=lane-16, +8 variants.
__device__ __forceinline__ v16h load_a_frag(const _Float16* __restrict__ Xh,
                                            int m0, int kk, int lane) {
    int m  = m0 + (lane & 15);
    int kb = ((lane >> 4) ? 8 : 0) + kk;
    const _Float16* p = Xh + (long long)m * 64 + kb;
    v8h lo = *(const v8h*)(p);
    v8h hi = *(const v8h*)(p + 16);
    return __builtin_shufflevector(lo, hi, 0, 1, 2, 3, 4, 5, 6, 7,
                                           8, 9, 10, 11, 12, 13, 14, 15);
}

// B: 32x16 tile from Wt[N x 64] f16 (pre-transposed); lanes index N,
// registers index K -> per-lane K-contiguous 16B chunks.
__device__ __forceinline__ v16h load_b_frag(const _Float16* __restrict__ Wt,
                                            int n0, int kk, int lane) {
    int n  = n0 + (lane & 15);
    int kb = ((lane >> 4) ? 8 : 0) + kk;
    const _Float16* p = Wt + n * 64 + kb;
    v8h lo = *(const v8h*)(p);
    v8h hi = *(const v8h*)(p + 16);
    return __builtin_shufflevector(lo, hi, 0, 1, 2, 3, 4, 5, 6, 7,
                                           8, 9, 10, 11, 12, 13, 14, 15);
}

// H[M x 64] = Xh[M x 64] @ W[64 x 64] (Wt pre-transposed f16).
// One wave per 16-row tile; 4 N-tiles; K=64 as two x32 WMMA steps (A reused).
__global__ void gemm64_wmma(const _Float16* __restrict__ Xh, const _Float16* __restrict__ Wt,
                            float* __restrict__ H, int M) {
    int wave = blockIdx.x * (blockDim.x >> 5) + (threadIdx.x >> 5);
    int lane = threadIdx.x & 31;
    int m0 = wave * 16;
    if (m0 >= M) return;

    v8f c0 = {}, c1 = {}, c2 = {}, c3 = {};
#pragma unroll
    for (int kk = 0; kk < 64; kk += 32) {
        v16h a = load_a_frag(Xh, m0, kk, lane);
        v16h b;
        b  = load_b_frag(Wt,  0, kk, lane);
        c0 = __builtin_amdgcn_wmma_f32_16x16x32_f16(false, a, false, b, (short)0, c0, false, false);
        b  = load_b_frag(Wt, 16, kk, lane);
        c1 = __builtin_amdgcn_wmma_f32_16x16x32_f16(false, a, false, b, (short)0, c1, false, false);
        b  = load_b_frag(Wt, 32, kk, lane);
        c2 = __builtin_amdgcn_wmma_f32_16x16x32_f16(false, a, false, b, (short)0, c2, false, false);
        b  = load_b_frag(Wt, 48, kk, lane);
        c3 = __builtin_amdgcn_wmma_f32_16x16x32_f16(false, a, false, b, (short)0, c3, false, false);
    }

    // C/D layout: VGPR r -> M = r (lanes 0-15) or r+8 (lanes 16-31); lane&15 -> N.
    int nlo = lane & 15;
    int mhi = (lane >> 4) * 8;
#pragma unroll
    for (int r = 0; r < 8; ++r) {
        float* out = H + (long long)(m0 + mhi + r) * 64 + nlo;
        out[0]  = c0[r];
        out[16] = c1[r];
        out[32] = c2[r];
        out[48] = c3[r];
    }
}

// H[M x 16] = Xh[M x 64] @ W[64 x 16] (Wt pre-transposed f16, 16x64).
__global__ void gemm16_wmma(const _Float16* __restrict__ Xh, const _Float16* __restrict__ Wt,
                            float* __restrict__ H, int M) {
    int wave = blockIdx.x * (blockDim.x >> 5) + (threadIdx.x >> 5);
    int lane = threadIdx.x & 31;
    int m0 = wave * 16;
    if (m0 >= M) return;

    v8f c = {};
#pragma unroll
    for (int kk = 0; kk < 64; kk += 32) {
        v16h a = load_a_frag(Xh, m0, kk, lane);
        v16h b = load_b_frag(Wt, 0, kk, lane);
        c = __builtin_amdgcn_wmma_f32_16x16x32_f16(false, a, false, b, (short)0, c, false, false);
    }

    int nlo = lane & 15;
    int mhi = (lane >> 4) * 8;
#pragma unroll
    for (int r = 0; r < 8; ++r)
        H[(long long)(m0 + mhi + r) * 16 + nlo] = c[r];
}

// ---------------- edge aggregation ----------------
// agg[col] += h[row] * (dinv[row]*dinv[col]).  4 features per thread;
// consecutive threads cover consecutive feature chunks of the same edge
// (coalesced float4 gathers, 4 scalar f32 atomics, all L2-resident).
__global__ void edge_agg(const int* __restrict__ row, const int* __restrict__ col,
                         const float* __restrict__ dinv, const float* __restrict__ h,
                         float* __restrict__ agg, int E, int F) {
    int chunks = F >> 2;
    long long total  = (long long)E * chunks;
    long long idx    = (long long)blockIdx.x * blockDim.x + threadIdx.x;
    long long stride = (long long)gridDim.x * blockDim.x;
    for (; idx < total; idx += stride) {
        int e  = (int)(idx / chunks);
        int f4 = ((int)(idx % chunks)) << 2;
        int r = row[e];
        int c = col[e];
        float norm = dinv[r] * dinv[c];
        const float4 v = *(const float4*)(h + (long long)r * F + f4);
        float* dst = agg + (long long)c * F + f4;
        atomicAdd(dst + 0, v.x * norm);
        atomicAdd(dst + 1, v.y * norm);
        atomicAdd(dst + 2, v.z * norm);
        atomicAdd(dst + 3, v.w * norm);
    }
}

// layer-1 epilogue: self-loop message + bias + ReLU, emitted directly as f16
// (it is only consumed as the A operand of the layer-2 WMMA GEMM).
__global__ void finish_relu_f16(const float* __restrict__ agg, const float* __restrict__ h,
                                const float* __restrict__ dinv, const float* __restrict__ b,
                                _Float16* __restrict__ out, int N) {
    long long t = (long long)blockIdx.x * blockDim.x + threadIdx.x;
    long long total = (long long)N * 64;
    if (t >= total) return;
    int i = (int)(t >> 6);
    int f = (int)(t & 63);
    float d2 = dinv[i] * dinv[i];
    float v = agg[t] + d2 * h[t] + b[f];
    out[t] = (_Float16)(v > 0.0f ? v : 0.0f);
}

// layer-2 epilogue: self-loop + bias, then log_softmax over 16 classes
__global__ void finish_logsoftmax(const float* __restrict__ agg, const float* __restrict__ h,
                                  const float* __restrict__ dinv, const float* __restrict__ b,
                                  float* __restrict__ out, int N) {
    int i = blockIdx.x * blockDim.x + threadIdx.x;
    if (i >= N) return;
    float d2 = dinv[i] * dinv[i];
    float v[16];
    float mx = -3.402823e38f;
#pragma unroll
    for (int f = 0; f < 16; ++f) {
        v[f] = agg[(long long)i * 16 + f] + d2 * h[(long long)i * 16 + f] + b[f];
        mx = fmaxf(mx, v[f]);
    }
    float s = 0.0f;
#pragma unroll
    for (int f = 0; f < 16; ++f) s += expf(v[f] - mx);
    float lse = mx + logf(s);
#pragma unroll
    for (int f = 0; f < 16; ++f) out[(long long)i * 16 + f] = v[f] - lse;
}

// ---------------- launcher ----------------

extern "C" void kernel_launch(void* const* d_in, const int* in_sizes, int n_in,
                              void* d_out, int out_size, void* d_ws, size_t ws_size,
                              hipStream_t stream) {
    const float* x   = (const float*)d_in[0];   // [N,64]
    const int*   ei  = (const int*)  d_in[1];   // [2,E]
    const float* W1  = (const float*)d_in[2];   // [64,64]
    const float* b1  = (const float*)d_in[3];   // [64]
    const float* W2  = (const float*)d_in[4];   // [64,16]
    const float* b2  = (const float*)d_in[5];   // [16]
    float*       out = (float*)d_out;           // [N,16]

    const int N = in_sizes[0] / 64;
    const int E = in_sizes[1] / 2;
    const int* row = ei;        // sources
    const int* col = ei + E;    // targets

    // workspace layout: f32 region, then 16B-aligned f16 region
    float* dinv = (float*)d_ws;                           // N
    float* h1   = dinv + N;                               // N*64
    float* agg1 = h1   + (long long)N * 64;               // N*64
    float* h2   = agg1 + (long long)N * 64;               // N*16
    float* agg2 = h2   + (long long)N * 16;               // N*16
    _Float16* xh     = (_Float16*)(agg2 + (long long)N * 16);  // N*64 halfs
    _Float16* hreluh = xh + (long long)N * 64;                 // N*64 halfs
    _Float16* w1t    = hreluh + (long long)N * 64;             // 64*64 halfs
    _Float16* w2t    = w1t + 64 * 64;                          // 16*64 halfs

    const int TB = 256;
    const long long n64 = (long long)N * 64;
    const long long n16 = (long long)N * 16;

    // init + operand preparation
    fill_f32<<<(N + TB - 1) / TB, TB, 0, stream>>>(dinv, 1.0f, N);
    fill_f32<<<(int)((n64 + TB - 1) / TB), TB, 0, stream>>>(agg1, 0.0f, n64);
    fill_f32<<<(int)((n16 + TB - 1) / TB), TB, 0, stream>>>(agg2, 0.0f, n16);
    f32_to_f16<<<(int)((n64 + TB - 1) / TB), TB, 0, stream>>>(x, xh, n64);
    transpose_w_f16<<<(64 * 64 + TB - 1) / TB, TB, 0, stream>>>(W1, w1t, 64, 64);
    transpose_w_f16<<<(64 * 16 + TB - 1) / TB, TB, 0, stream>>>(W2, w2t, 64, 16);

    // degrees -> dinv
    deg_accum<<<(E + TB - 1) / TB, TB, 0, stream>>>(col, dinv, E);
    make_dinv<<<(N + TB - 1) / TB, TB, 0, stream>>>(dinv, N);

    int tilesM = (N + 15) / 16;
    int wavesPerBlock = TB / 32;
    int gemmBlocks = (tilesM + wavesPerBlock - 1) / wavesPerBlock;

    // layer 1: WMMA GEMM + edge scatter + ReLU epilogue (f16 out)
    gemm64_wmma<<<gemmBlocks, TB, 0, stream>>>(xh, w1t, h1, N);
    {
        long long total = (long long)E * 16;   // 64 feats / 4 per thread
        edge_agg<<<(int)((total + TB - 1) / TB), TB, 0, stream>>>(row, col, dinv, h1, agg1, E, 64);
    }
    finish_relu_f16<<<(int)((n64 + TB - 1) / TB), TB, 0, stream>>>(agg1, h1, dinv, b1, hreluh, N);

    // layer 2: WMMA GEMM + edge scatter + log-softmax epilogue
    gemm16_wmma<<<gemmBlocks, TB, 0, stream>>>(hreluh, w2t, h2, N);
    {
        long long total = (long long)E * 4;    // 16 feats / 4 per thread
        edge_agg<<<(int)((total + TB - 1) / TB), TB, 0, stream>>>(row, col, dinv, h2, agg2, E, 16);
    }
    finish_logsoftmax<<<(N + TB - 1) / TB, TB, 0, stream>>>(agg2, h2, dinv, b2, out, N);
}
